// AttentionModel_55198919688262
// MI455X (gfx1250) — compile-verified, matrix operand order checked by reading, not run on hardware
//
#include <hip/hip_runtime.h>
#include <hip/hip_bf16.h>
#include <math.h>

// Problem constants (match reference)
#define B_   128
#define T_   64
#define D_   300
#define DP_  320          // D padded to multiple of 32 for WMMA K
#define H_   512
#define G4_  2048         // 4*H
#define BT_  (B_ * T_)    // 8192

typedef __attribute__((ext_vector_type(16))) _Float16 v16h;
typedef __attribute__((ext_vector_type(8)))  _Float16 v8h;
typedef __attribute__((ext_vector_type(8)))  float    v8f;

__device__ __forceinline__ float sigmoidf_(float x) { return 1.0f / (1.0f + __expf(-x)); }

// ---------------------------------------------------------------------------
// CDNA5 async global->LDS copies (ASYNCcnt path, ISA ch.10 / 15.18).
// lds  = LDS byte address (low 32 bits of generic pointer == LDS offset)
// gsrc = per-lane 64-bit global address
// ---------------------------------------------------------------------------
__device__ __forceinline__ void async_g2l_b128(unsigned lds, const void* gsrc)
{
    asm volatile("global_load_async_to_lds_b128 %0, %1, off"
                 :: "v"(lds), "v"(gsrc) : "memory");
}
__device__ __forceinline__ void async_g2l_b32(unsigned lds, const void* gsrc)
{
    asm volatile("global_load_async_to_lds_b32 %0, %1, off"
                 :: "v"(lds), "v"(gsrc) : "memory");
}
__device__ __forceinline__ void wait_async0()
{
    asm volatile("s_wait_asynccnt 0" ::: "memory");
}

// ---------------------------------------------------------------------------
// Embedding gather: tokens -> f16 rows [BT, DP_], zero-padded cols D_..DP_-1
// ---------------------------------------------------------------------------
__global__ void __launch_bounds__(256)
embed_cast(const int* __restrict__ tok, const float* __restrict__ emb,
           _Float16* __restrict__ outp)
{
    int row = blockIdx.x;                       // 0..BT-1
    long src = (long)tok[row] * D_;
    for (int d = threadIdx.x; d < DP_; d += 256)
        outp[(long)row * DP_ + d] = (d < D_) ? (_Float16)emb[src + d] : (_Float16)0.0f;
}

// ---------------------------------------------------------------------------
// Pack an fp32 row-major [Krows, N] weight (ld = ldw) into f16 WMMA B-fragment
// order. Packed index: ((kt*ntB + nt)*32 + lane)*16 + e, where element e of
// lane (g = lane>>4, n = lane&15) holds B[kt*32 + (e<8?0:16) + g*8 + (e&7)][nt*16+n].
// Rows >= Krows are zero (K padding).
// ---------------------------------------------------------------------------
__global__ void __launch_bounds__(256)
pack_b(const float* __restrict__ W, int ldw, int Krows,
       _Float16* __restrict__ outp, int N, int total)
{
    int idx = blockIdx.x * blockDim.x + threadIdx.x;
    if (idx >= total) return;
    int e    = idx & 15;
    int lane = (idx >> 4) & 31;
    int tile = idx >> 9;
    int ntB  = N >> 4;
    int kt   = tile / ntB;
    int nt   = tile - kt * ntB;
    int g    = lane >> 4;
    int n    = lane & 15;
    int k    = kt * 32 + ((e < 8) ? 0 : 16) + g * 8 + (e & 7);
    int col  = nt * 16 + n;
    outp[idx] = (k < Krows) ? (_Float16)W[(long)k * ldw + col] : (_Float16)0.0f;
}

// ---------------------------------------------------------------------------
// WMMA GEMM, double-buffered async-LDS staging.
// C[M,N](f32) = A[M,K](f16 row-major, lda) x Bpacked (+ Cinit + bias).
// flags bit0: accumulate into existing C;  bit1: apply tanh at the end.
//
// Block = 8 waves computes one 16x256 macro-tile (1 M-tile x 16 N-tiles);
// wave w owns N-tiles ntBase+w and ntBase+w+8 (2 WMMA accumulators).
// Per 32-wide K step the block stages into LDS buffer kt&1:
//   sA: 16x32 f16 A slab  (1 KB)  -- one async b32 per thread (uniform EXEC)
//   sB: 16 packed B tiles (16 KB) -- four async b128 per thread
// Pipeline: stage(kt+1) is issued right after the top-of-iteration barrier,
// so its async copy overlaps the WMMAs on buffer kt&1. One barrier per step:
// reaching it proves (a) this wave's s_wait_asynccnt drained stage kt and
// (b) every wave finished reading the buffer stage kt+1 will overwrite.
// Requires Ntiles % 16 == 0 (call sites: 128, 32).
// ---------------------------------------------------------------------------
__global__ void __launch_bounds__(256)
gemm_wmma_f16(const _Float16* __restrict__ A, int lda,
              const _Float16* __restrict__ Bp, int ntB,
              float* __restrict__ C, int ldc,
              const float* __restrict__ Cinit, int ldci,
              const float* __restrict__ bias,
              int Mtiles, int Ntiles, int Ktiles, int flags)
{
    __shared__ __align__(16) _Float16 sA[2][16 * 32];    // 2 x 1 KB
    __shared__ __align__(16) _Float16 sB[2][16 * 512];   // 2 x 16 KB

    int tid  = threadIdx.x;
    int wave = tid >> 5;
    int lane = tid & 31;
    int nblk = Ntiles >> 4;
    int mt     = blockIdx.x / nblk;
    int ntBase = (blockIdx.x - mt * nblk) << 4;
    if (mt >= Mtiles) return;
    int g = lane >> 4;        // K-half selector for fragments
    int l = lane & 15;        // row (A) / col (B,C)

    unsigned sA_b = (unsigned)(size_t)&sA[0][0];
    unsigned sB_b = (unsigned)(size_t)&sB[0][0];
    const _Float16* Abase = A + (long)(mt * 16) * lda;   // 16 rows of A
    int arow = tid >> 4, awrd = tid & 15;                // A-copy mapping

    auto stage = [&](int kt, int buf) {
        // B slab: 8192 f16, contiguous in packed layout; 32 f16 per thread.
        const _Float16* bsrc = Bp + (((long)kt * ntB + ntBase) << 9);
        unsigned sbb = sB_b + (unsigned)buf * 16384u + (unsigned)tid * 64u;
        async_g2l_b128(sbb,      bsrc + tid * 32);
        async_g2l_b128(sbb + 16, bsrc + tid * 32 + 8);
        async_g2l_b128(sbb + 32, bsrc + tid * 32 + 16);
        async_g2l_b128(sbb + 48, bsrc + tid * 32 + 24);
        // A slab: 16 rows x 32 f16 = 1 KB; one dword per thread.
        async_g2l_b32(sA_b + (unsigned)buf * 1024u + (unsigned)(arow * 64 + awrd * 4),
                      Abase + (long)arow * lda + kt * 32 + awrd * 2);
    };

    v8f acc0 = {};
    v8f acc1 = {};
    stage(0, 0);
    for (int kt = 0; kt < Ktiles; ++kt) {
        int cur = kt & 1;
        wait_async0();          // drain stage kt (only it is outstanding here)
        __syncthreads();
        if (kt + 1 < Ktiles) stage(kt + 1, cur ^ 1);   // overlaps compute below

        const _Float16* ap = &sA[cur][l * 32 + g * 8];
        v8h alo = *(const v8h*)(ap);        // K = kt*32 + g*8 .. +7
        v8h ahi = *(const v8h*)(ap + 16);   // K = kt*32 + 16 + g*8 .. +7
        v16h a;
#pragma unroll
        for (int i = 0; i < 8; ++i) { a[i] = alo[i]; a[8 + i] = ahi[i]; }

        const _Float16* bp0 = &sB[cur][wave * 512 + lane * 16];
        v8h b0lo = *(const v8h*)(bp0);
        v8h b0hi = *(const v8h*)(bp0 + 8);
        v16h b0;
#pragma unroll
        for (int i = 0; i < 8; ++i) { b0[i] = b0lo[i]; b0[8 + i] = b0hi[i]; }
        const _Float16* bp1 = &sB[cur][(wave + 8) * 512 + lane * 16];
        v8h b1lo = *(const v8h*)(bp1);
        v8h b1hi = *(const v8h*)(bp1 + 8);
        v16h b1;
#pragma unroll
        for (int i = 0; i < 8; ++i) { b1[i] = b1lo[i]; b1[8 + i] = b1hi[i]; }

        acc0 = __builtin_amdgcn_wmma_f32_16x16x32_f16(false, a, false, b0,
                                                      (short)0, acc0, false, false);
        acc1 = __builtin_amdgcn_wmma_f32_16x16x32_f16(false, a, false, b1,
                                                      (short)0, acc1, false, false);
    }

    int col0 = (ntBase + wave) * 16 + l;
    int col1 = col0 + 128;                 // (+8 N-tiles)
#pragma unroll
    for (int r = 0; r < 8; ++r) {
        int row  = mt * 16 + g * 8 + r;    // C/D layout: vgpr r -> M=r / M=r+8
        long rowb = (long)row * ldc;
        float v0 = acc0[r];
        float v1 = acc1[r];
        if (Cinit) {
            long ci = (long)row * ldci;
            v0 += Cinit[ci + col0];
            v1 += Cinit[ci + col1];
        }
        if (bias) { v0 += bias[col0]; v1 += bias[col1]; }
        if (flags & 1) { v0 += C[rowb + col0]; v1 += C[rowb + col1]; }
        if (flags & 2) { v0 = tanhf(v0); v1 = tanhf(v1); }
        C[rowb + col0] = v0;
        C[rowb + col1] = v1;
    }
}

// ---------------------------------------------------------------------------
// LSTM pointwise step: gates[B,4H] -> update c,h with sequence-length mask,
// emit Y[:,t,:] (f32 and f16) and h (f32 and f16).
// Gate order matches jnp.split: i, j, f, o. FORGET_BIAS = 1.0.
// ---------------------------------------------------------------------------
__global__ void __launch_bounds__(256)
lstm_point(const float* __restrict__ gates, float* __restrict__ c,
           float* __restrict__ h, _Float16* __restrict__ h16,
           float* __restrict__ Yf, _Float16* __restrict__ Yh,
           const int* __restrict__ seqlen, int t)
{
    int b = blockIdx.x;
    float v = (t < seqlen[b]) ? 1.0f : 0.0f;
    long gbase = (long)b * G4_;
    for (int k = threadIdx.x; k < H_; k += 256) {
        float gi = gates[gbase + k];
        float gj = gates[gbase + H_ + k];
        float gf = gates[gbase + 2 * H_ + k];
        float go = gates[gbase + 3 * H_ + k];
        long idx = (long)b * H_ + k;
        float cold = c[idx], hold = h[idx];
        float cn = cold * sigmoidf_(gf + 1.0f) + sigmoidf_(gi) * tanhf(gj);
        float hn = tanhf(cn) * sigmoidf_(go);
        float cu = v * cn + (1.0f - v) * cold;
        float hu = v * hn + (1.0f - v) * hold;
        c[idx] = cu;
        h[idx] = hu;
        h16[idx] = (_Float16)hu;
        long yidx = ((long)b * T_ + t) * H_ + k;
        float yv = v * hn;
        Yf[yidx] = yv;
        Yh[yidx] = (_Float16)yv;
    }
}

// ---------------------------------------------------------------------------
// Attention step: score -> softmax -> Y·alpha -> r update.
// One block per batch row. u = h_t@Wh + r@Wr (precomputed), q = r@Wt.
// ---------------------------------------------------------------------------
__global__ void __launch_bounds__(256)
att_update(const float* __restrict__ WyY, const float* __restrict__ u,
           const float* __restrict__ q, const float* __restrict__ wv,
           const float* __restrict__ Y,
           const int* __restrict__ seq_mask, const int* __restrict__ seq_last,
           float* __restrict__ r, _Float16* __restrict__ r16,
           float* __restrict__ rL, _Float16* __restrict__ rL16, int t)
{
    __shared__ float s_u[H_];
    __shared__ float s_sc[T_];
    __shared__ float s_inv;
    int b = blockIdx.x;
    int tid = threadIdx.x;
    for (int hh = tid; hh < H_; hh += 256) s_u[hh] = u[b * H_ + hh];
    __syncthreads();

    int wave = tid >> 5, lane = tid & 31;
    int slen = seq_mask[b];
    for (int tp = wave; tp < T_; tp += 8) {
        const float* wy = WyY + ((long)b * T_ + tp) * H_;
        float s = 0.0f;
        for (int hh = lane; hh < H_; hh += 32)
            s += tanhf(wy[hh] + s_u[hh]) * wv[hh];
        for (int o = 16; o > 0; o >>= 1) s += __shfl_down(s, o, 32);
        if (lane == 0) s_sc[tp] = s + ((tp < slen) ? 0.0f : -10000.0f);
    }
    __syncthreads();

    if (tid == 0) {
        float mx = s_sc[0];
        for (int i = 1; i < T_; ++i) mx = fmaxf(mx, s_sc[i]);
        float ssum = 0.0f;
        for (int i = 0; i < T_; ++i) {
            float e = __expf(s_sc[i] - mx);
            s_sc[i] = e;
            ssum += e;
        }
        s_inv = 1.0f / ssum;
    }
    __syncthreads();

    float inv = s_inv;
    int last = (t == seq_last[b] - 1);
    for (int hh = tid; hh < H_; hh += 256) {
        float acc = 0.0f;
        const float* yb = Y + (long)b * T_ * H_ + hh;
        for (int tp = 0; tp < T_; ++tp) acc += yb[(long)tp * H_] * s_sc[tp];
        acc *= inv;
        float rn = acc + tanhf(q[b * H_ + hh]);
        long idx = (long)b * H_ + hh;
        r[idx]  = rn;
        r16[idx] = (_Float16)rn;
        if (last) { rL[idx] = rn; rL16[idx] = (_Float16)rn; }
    }
}

// ---------------------------------------------------------------------------
// Final: out[b, j] = sum_h (la+lb)[b,h] * U[h,j] + bU[j], j in {0,1}
// ---------------------------------------------------------------------------
__global__ void __launch_bounds__(64)
final_out_k(const float* __restrict__ za, const float* __restrict__ zb,
            const float* __restrict__ U, const float* __restrict__ bU,
            float* __restrict__ out)
{
    int b = blockIdx.x;
    int j = threadIdx.x >> 5, lane = threadIdx.x & 31;
    float s = 0.0f;
    for (int hh = lane; hh < H_; hh += 32)
        s += (za[b * H_ + hh] + zb[b * H_ + hh]) * U[hh * 2 + j];
    for (int o = 16; o > 0; o >>= 1) s += __shfl_down(s, o, 32);
    if (lane == 0) out[b * 2 + j] = s + bU[j];
}

__global__ void zero_words(unsigned int* __restrict__ p, int n)
{
    int i = blockIdx.x * blockDim.x + threadIdx.x;
    if (i < n) p[i] = 0u;
}

// ---------------------------------------------------------------------------
extern "C" void kernel_launch(void* const* d_in, const int* in_sizes, int n_in,
                              void* d_out, int out_size, void* d_ws, size_t ws_size,
                              hipStream_t stream)
{
    (void)in_sizes; (void)n_in; (void)out_size; (void)ws_size;

    const int*   tokens1 = (const int*)d_in[0];
    const int*   tokens2 = (const int*)d_in[1];
    const int*   s1      = (const int*)d_in[2];
    const int*   s2      = (const int*)d_in[3];
    const float* emb     = (const float*)d_in[4];
    const float* W1      = (const float*)d_in[5];
    const float* b1      = (const float*)d_in[6];
    const float* W2      = (const float*)d_in[7];
    const float* b2      = (const float*)d_in[8];
    const float* Wy      = (const float*)d_in[9];
    const float* Wh      = (const float*)d_in[10];
    const float* Wr      = (const float*)d_in[11];
    const float* wv      = (const float*)d_in[12];
    const float* Wt      = (const float*)d_in[13];
    const float* Wp      = (const float*)d_in[14];
    const float* Wx      = (const float*)d_in[15];
    const float* U       = (const float*)d_in[16];
    const float* bU      = (const float*)d_in[17];
    float* out = (float*)d_out;

    // ---- workspace bump allocator (256B aligned) ----
    char* ws = (char*)d_ws;
    size_t off = 0;
    auto alloc = [&](size_t bytes) -> void* {
        void* p = ws + off;
        off = (off + bytes + 255) & ~(size_t)255;
        return p;
    };

    _Float16* e1h  = (_Float16*)alloc((size_t)BT_ * DP_ * 2);
    _Float16* e2h  = (_Float16*)alloc((size_t)BT_ * DP_ * 2);
    _Float16* W1xp = (_Float16*)alloc((size_t)DP_ * G4_ * 2);
    _Float16* W2xp = (_Float16*)alloc((size_t)DP_ * G4_ * 2);
    _Float16* W1hp = (_Float16*)alloc((size_t)H_ * G4_ * 2);
    _Float16* W2hp = (_Float16*)alloc((size_t)H_ * G4_ * 2);
    _Float16* Wyp  = (_Float16*)alloc((size_t)H_ * H_ * 2);
    _Float16* Whp  = (_Float16*)alloc((size_t)H_ * H_ * 2);
    _Float16* Wrp  = (_Float16*)alloc((size_t)H_ * H_ * 2);
    _Float16* Wtp  = (_Float16*)alloc((size_t)H_ * H_ * 2);
    _Float16* Wpp  = (_Float16*)alloc((size_t)H_ * H_ * 2);
    _Float16* Wxap = (_Float16*)alloc((size_t)H_ * H_ * 2);

    float*    Gx[4];  float* Yf[4];  _Float16* Yh[4];
    float*    cst[4]; float* hst[4]; _Float16* hh16[4];
    float*    gate[4];
    for (int r = 0; r < 4; ++r) {
        Gx[r]   = (float*)alloc((size_t)BT_ * G4_ * 4);
        Yf[r]   = (float*)alloc((size_t)BT_ * H_ * 4);
        Yh[r]   = (_Float16*)alloc((size_t)BT_ * H_ * 2);
        cst[r]  = (float*)alloc((size_t)B_ * H_ * 4);
        hst[r]  = (float*)alloc((size_t)B_ * H_ * 4);
        hh16[r] = (_Float16*)alloc((size_t)B_ * H_ * 2);
        gate[r] = (float*)alloc((size_t)B_ * G4_ * 4);
    }
    float* WyYb[2]; float* ub[2]; float* qb[2]; float* rb[2];
    _Float16* rh16[2]; float* rLb[2]; _Float16* rLh16[2]; float* zb2[2];
    for (int a = 0; a < 2; ++a) {
        WyYb[a]  = (float*)alloc((size_t)BT_ * H_ * 4);
        ub[a]    = (float*)alloc((size_t)B_ * H_ * 4);
        qb[a]    = (float*)alloc((size_t)B_ * H_ * 4);
        rb[a]    = (float*)alloc((size_t)B_ * H_ * 4);
        rh16[a]  = (_Float16*)alloc((size_t)B_ * H_ * 2);
        rLb[a]   = (float*)alloc((size_t)B_ * H_ * 4);
        rLh16[a] = (_Float16*)alloc((size_t)B_ * H_ * 2);
        zb2[a]   = (float*)alloc((size_t)B_ * H_ * 4);
    }

    auto zero = [&](void* p, size_t bytes) {
        int n = (int)(bytes / 4);
        zero_words<<<(n + 255) / 256, 256, 0, stream>>>((unsigned int*)p, n);
    };
    auto gemm = [&](const _Float16* A, int lda, const _Float16* Bp,
                    float* C, int ldc, const float* Ci, int ldci,
                    const float* bias, int Mt, int Nt, int Kt, int fl) {
        int blocks = Mt * (Nt >> 4);   // Nt % 16 == 0 at every call site
        gemm_wmma_f16<<<blocks, 256, 0, stream>>>(A, lda, Bp, Nt, C, ldc,
                                                  Ci, ldci, bias, Mt, Nt, Kt, fl);
    };
    auto pack = [&](const float* W, int ldw, int kr, _Float16* outp, int kpad, int n) {
        int total = kpad * n;
        pack_b<<<(total + 255) / 256, 256, 0, stream>>>(W, ldw, kr, outp, n, total);
    };

    // ---- stage 0: embeddings + weight packing ----
    embed_cast<<<BT_, 256, 0, stream>>>(tokens1, emb, e1h);
    embed_cast<<<BT_, 256, 0, stream>>>(tokens2, emb, e2h);
    pack(W1,             G4_, D_, W1xp, DP_, G4_);   // W1 rows 0..D-1   (x part)
    pack(W1 + D_ * G4_,  G4_, H_, W1hp, H_,  G4_);   // W1 rows D..D+H-1 (h part)
    pack(W2,             G4_, D_, W2xp, DP_, G4_);
    pack(W2 + D_ * G4_,  G4_, H_, W2hp, H_,  G4_);
    pack(Wy, H_, H_, Wyp,  H_, H_);
    pack(Wh, H_, H_, Whp,  H_, H_);
    pack(Wr, H_, H_, Wrp,  H_, H_);
    pack(Wt, H_, H_, Wtp,  H_, H_);
    pack(Wp, H_, H_, Wpp,  H_, H_);
    pack(Wx, H_, H_, Wxap, H_, H_);

    // ---- stage 1: input GEMMs  Gx = x @ Wx + b   (4 runs) ----
    // run0: (e1,W1,b1,s1)  run1: (e2,W2,b2,s2)  run2: (e2,W1,b1,s2)  run3: (e1,W2,b2,s1)
    const _Float16* Aruns[4] = { e1h, e2h, e2h, e1h };
    const _Float16* Wxr[4]   = { W1xp, W2xp, W1xp, W2xp };
    const _Float16* Whr[4]   = { W1hp, W2hp, W1hp, W2hp };
    const float*    biasr[4] = { b1, b2, b1, b2 };
    const int*      seqr[4]  = { s1, s2, s2, s1 };
    for (int r = 0; r < 4; ++r) {
        gemm(Aruns[r], DP_, Wxr[r], Gx[r], G4_, nullptr, 0, biasr[r],
             BT_ / 16, G4_ / 16, DP_ / 32, 0);
        zero(cst[r], (size_t)B_ * H_ * 4);
        zero(hst[r], (size_t)B_ * H_ * 4);
        zero(hh16[r], (size_t)B_ * H_ * 2);
    }

    // ---- stage 2: LSTM recurrence over T ----
    for (int t = 0; t < T_; ++t) {
        for (int r = 0; r < 4; ++r) {
            gemm(hh16[r], H_, Whr[r], gate[r], G4_,
                 Gx[r] + (size_t)t * G4_, T_ * G4_, nullptr,
                 B_ / 16, G4_ / 16, H_ / 32, 0);
            lstm_point<<<B_, 256, 0, stream>>>(gate[r], cst[r], hst[r], hh16[r],
                                               Yf[r], Yh[r], seqr[r], t);
        }
    }

    // ---- stage 3: WyY = Y @ Wy ----
    gemm(Yh[0], H_, Wyp, WyYb[0], H_, nullptr, 0, nullptr,
         BT_ / 16, H_ / 16, H_ / 32, 0);
    gemm(Yh[2], H_, Wyp, WyYb[1], H_, nullptr, 0, nullptr,
         BT_ / 16, H_ / 16, H_ / 32, 0);

    // ---- stage 4: attention scan per asym ----
    // la: Y=run0, Y2/last_h=run1, mask=s1, oh uses s2
    // lb: Y=run2, Y2/last_h=run3, mask=s2, oh uses s1
    for (int a = 0; a < 2; ++a) {
        const _Float16* Y2h   = (a == 0) ? Yh[1] : Yh[3];
        const float*    Yfa   = (a == 0) ? Yf[0] : Yf[2];
        const int*      mseq  = (a == 0) ? s1 : s2;
        const int*      lseq  = (a == 0) ? s2 : s1;
        const _Float16* lastH = (a == 0) ? hh16[1] : hh16[3];

        zero(rb[a], (size_t)B_ * H_ * 4);
        zero(rh16[a], (size_t)B_ * H_ * 2);
        zero(rLb[a], (size_t)B_ * H_ * 4);
        zero(rLh16[a], (size_t)B_ * H_ * 2);

        for (int t = 0; t < T_; ++t) {
            // u = h_t @ Wh + r @ Wr
            gemm(Y2h + (size_t)t * H_, T_ * H_, Whp, ub[a], H_,
                 nullptr, 0, nullptr, B_ / 16, H_ / 16, H_ / 32, 0);
            gemm(rh16[a], H_, Wrp, ub[a], H_,
                 nullptr, 0, nullptr, B_ / 16, H_ / 16, H_ / 32, 1);
            // q = r @ Wt
            gemm(rh16[a], H_, Wtp, qb[a], H_,
                 nullptr, 0, nullptr, B_ / 16, H_ / 16, H_ / 32, 0);
            att_update<<<B_, 256, 0, stream>>>(WyYb[a], ub[a], qb[a], wv, Yfa,
                                               mseq, lseq, rb[a], rh16[a],
                                               rLb[a], rLh16[a], t);
        }
        // z = tanh(r_L @ Wp + last_h @ Wx)
        gemm(rLh16[a], H_, Wpp, zb2[a], H_,
             nullptr, 0, nullptr, B_ / 16, H_ / 16, H_ / 32, 0);
        gemm(lastH, H_, Wxap, zb2[a], H_,
             nullptr, 0, nullptr, B_ / 16, H_ / 16, H_ / 32, 3 /*ACC|TANH*/);
    }

    // ---- stage 5: output projection ----
    final_out_k<<<B_, 64, 0, stream>>>(zb2[0], zb2[1], U, bU, out);
}
